// ESNLayer_10273561772227
// MI455X (gfx1250) — matrix operand order repaired and checked by reading, not compiled, not run on hardware
//
#include <hip/hip_runtime.h>
#include <stdint.h>

// Problem constants (match reference)
#define B_SZ 32
#define C_SZ 64
#define T_SZ 2048
#define H_SZ 1024
#define LEAK 0.9f

// Mapping
#define NWG 16                 // workgroups; each owns H/NWG = 64 output columns
#define COLS_PER_WG (H_SZ / NWG)
#define NTHREADS 256           // 8 waves of 32

typedef __attribute__((ext_vector_type(16))) __bf16 v16bf;
typedef __attribute__((ext_vector_type(8)))  float  v8f;

union V16 { v16bf v; uint4 q[2]; };

__device__ __forceinline__ unsigned short f2bf(float f) {
    unsigned u = __float_as_uint(f);
    unsigned r = u + 0x7FFFu + ((u >> 16) & 1u);   // round-to-nearest-even
    return (unsigned short)(r >> 16);
}

// Branchless tanh: tanh(x) = sign(x) * (1-e)/(1+e), e = exp(-2|x|) = 2^(-2*log2(e)*|x|)
// Uses v_exp_f32 + v_rcp_f32 + bit-ops only: no EXEC-mask divergence next to WMMA results.
__device__ __forceinline__ float fast_tanh(float x) {
    const float NEG2_LOG2E = -2.885390081777927f;      // -2*log2(e)
    float a = __builtin_fabsf(x);
    float e = __builtin_amdgcn_exp2f(NEG2_LOG2E * a);  // exp(-2|x|) in (0,1]
    float t = (1.0f - e) * __builtin_amdgcn_rcpf(1.0f + e);
    unsigned ut = __float_as_uint(t);
    unsigned ux = __float_as_uint(x);
    return __uint_as_float((ut & 0x7FFFFFFFu) | (ux & 0x80000000u));
}

// ---------------- prep kernels ----------------

__global__ void cvt_bf16_kernel(const float* __restrict__ src,
                                unsigned short* __restrict__ dst, int n) {
    int i = blockIdx.x * blockDim.x + threadIdx.x;
    if (i < n) dst[i] = f2bf(src[i]);
}

// x: (B,C,T) fp32 -> xbf: (T,B,C) bf16  (coalesced reads along t)
__global__ void xpose_x_kernel(const float* __restrict__ x,
                               unsigned short* __restrict__ xbf) {
    int i = blockIdx.x * blockDim.x + threadIdx.x;
    if (i >= B_SZ * C_SZ * T_SZ) return;
    int t  = i % T_SZ;
    int bc = i / T_SZ;
    int c  = bc % C_SZ;
    int bb = bc / C_SZ;
    xbf[((size_t)t * B_SZ + bb) * C_SZ + c] = f2bf(x[i]);
}

__global__ void init_state_kernel(unsigned short* __restrict__ hbf,
                                  unsigned* __restrict__ cnt) {
    int i = blockIdx.x * blockDim.x + threadIdx.x;
    if (i == 0) *cnt = 0u;
    if (i < 2 * B_SZ * H_SZ) hbf[i] = 0;
}

// ---------------- persistent recurrence kernel ----------------
//
// Grid: NWG blocks x 256 threads. Each block owns 64 output columns;
// W_res slice (64x1024 bf16 = 128KB) + W_in slice (64x64 bf16 = 8KB) live in LDS.
// Wave w: mtile = w>>2 (batch rows 0-15 / 16-31), ntile = w&3 (16-col group).
// Per timestep per wave: 2 WMMAs (input proj) + 32 WMMAs (recurrence, K=H=1024)
// spread over 4 INDEPENDENT accumulator chains (removes the D->C serial hazard),
// then branchless tanh + leak (fp32 state in registers) + stores, then grid barrier.

__global__ void esn_recurrence_kernel(
    const unsigned short* __restrict__ xbf,     // (T,B,C) bf16
    const unsigned short* __restrict__ wres_bf, // (H,H) bf16 row-major (j,i): B[k][n]=W_res[n][k]
    const unsigned short* __restrict__ win_bf,  // (H,C) bf16 row-major
    const float* __restrict__ bias,             // (H) fp32
    unsigned short* __restrict__ hbf,           // (2,B,H) bf16 double buffer
    float* __restrict__ out,                    // (B,H,T) fp32
    unsigned* __restrict__ barrier_cnt)
{
    extern __shared__ unsigned short smem[];
    unsigned short* wres_lds = smem;                         // 64*1024 halves
    unsigned short* win_lds  = smem + COLS_PER_WG * H_SZ;    // 64*64 halves

    const int wg  = blockIdx.x;
    const int tid = threadIdx.x;
    const int colBase = wg * COLS_PER_WG;

    // Preload stationary weights into LDS (uint4 = 8 bf16 per transfer).
    {
        const uint4* src = (const uint4*)(wres_bf + (size_t)colBase * H_SZ);
        uint4* dst = (uint4*)wres_lds;
        const int n = COLS_PER_WG * H_SZ / 8;                // 8192
        for (int i = tid; i < n; i += NTHREADS) dst[i] = src[i];
        const uint4* src2 = (const uint4*)(win_bf + (size_t)colBase * C_SZ);
        uint4* dst2 = (uint4*)win_lds;
        const int n2 = COLS_PER_WG * C_SZ / 8;               // 512
        for (int i = tid; i < n2; i += NTHREADS) dst2[i] = src2[i];
    }
    __syncthreads();

    const int wave   = tid >> 5;
    const int lane   = tid & 31;
    const int laneLo = lane & 15;
    const int laneHi = (lane >> 4) & 1;
    const int mtile  = wave >> 2;              // 0..1 : batch tile
    const int ntile  = wave & 3;               // 0..3 : column tile in WG
    const int mbase  = mtile * 16;
    const int rloc   = ntile * 16 + laneLo;    // local W row == output column (B-operand N=lane&15)
    const int nglob  = colBase + rloc;
    const float bval = bias[nglob];

    // A-operand lane mapping (16-bit A 16x32): lanes 0-15 K={0..7,16..23}, lanes 16-31 K={8..15,24..31}
    const int m     = mbase + laneLo;          // A row for this lane
    const int kselA = laneHi * 8;
    // B-operand lane mapping (16-bit B 32x16): lanes 0-15 K={0..7,16..23}, lanes 16-31 K={8..15,24..31}
    const int kselB = laneHi * 8;

    const unsigned short* win_row  = win_lds  + (size_t)rloc * C_SZ;
    const unsigned short* wres_row = wres_lds + (size_t)rloc * H_SZ;

    // fp32 leak state for this wave's own output elements (h0 = 0):
    // acc[r] <-> element (M = mbase + r + 8*laneHi, N = nglob), same every step.
    float hstate[8];
    #pragma unroll
    for (int r = 0; r < 8; ++r) hstate[r] = 0.0f;

    for (int t = 0; t < T_SZ; ++t) {
        const int cur = t & 1;
        const unsigned short* hcur = hbf + (size_t)cur * (B_SZ * H_SZ);
        unsigned short*       hnxt = hbf + (size_t)(cur ^ 1) * (B_SZ * H_SZ);

        // 4 independent accumulator chains (bias folded into chain 0 only).
        v8f acc0, acc1, acc2, acc3;
        #pragma unroll
        for (int i = 0; i < 8; ++i) {
            acc0[i] = bval; acc1[i] = 0.0f; acc2[i] = 0.0f; acc3[i] = 0.0f;
        }

        // ---- input projection: xp_t tile += x_t (16x64) . W_in_sliceT (64x16); chains 0,1
        const unsigned short* xrow = xbf + ((size_t)t * B_SZ + m) * C_SZ;
        {
            V16 a0, a1, b0, b1;
            a0.q[0] = *(const uint4*)(xrow + kselA);
            a0.q[1] = *(const uint4*)(xrow + kselA + 16);
            a1.q[0] = *(const uint4*)(xrow + 32 + kselA);
            a1.q[1] = *(const uint4*)(xrow + 32 + kselA + 16);
            b0.q[0] = *(const uint4*)(win_row + kselB);
            b0.q[1] = *(const uint4*)(win_row + kselB + 16);
            b1.q[0] = *(const uint4*)(win_row + 32 + kselB);
            b1.q[1] = *(const uint4*)(win_row + 32 + kselB + 16);
            acc0 = __builtin_amdgcn_wmma_f32_16x16x32_bf16(
                false, a0.v, false, b0.v, (short)0, acc0, false, false);
            acc1 = __builtin_amdgcn_wmma_f32_16x16x32_bf16(
                false, a1.v, false, b1.v, (short)0, acc1, false, false);
        }

        // ---- recurrence: += h (16x1024) . W_res_sliceT (1024x16); 32 WMMAs, 4 chains
        const unsigned short* hrow = hcur + (size_t)m * H_SZ;
        #pragma unroll 2
        for (int kt = 0; kt < 32; kt += 4) {
            V16 a[4], bm[4];
            #pragma unroll
            for (int j = 0; j < 4; ++j) {
                int ka = (kt + j) * 32 + kselA;
                a[j].q[0] = *(const uint4*)(hrow + ka);
                a[j].q[1] = *(const uint4*)(hrow + ka + 16);
                int kb = (kt + j) * 32 + kselB;
                bm[j].q[0] = *(const uint4*)(wres_row + kb);
                bm[j].q[1] = *(const uint4*)(wres_row + kb + 16);
            }
            acc0 = __builtin_amdgcn_wmma_f32_16x16x32_bf16(
                false, a[0].v, false, bm[0].v, (short)0, acc0, false, false);
            acc1 = __builtin_amdgcn_wmma_f32_16x16x32_bf16(
                false, a[1].v, false, bm[1].v, (short)0, acc1, false, false);
            acc2 = __builtin_amdgcn_wmma_f32_16x16x32_bf16(
                false, a[2].v, false, bm[2].v, (short)0, acc2, false, false);
            acc3 = __builtin_amdgcn_wmma_f32_16x16x32_bf16(
                false, a[3].v, false, bm[3].v, (short)0, acc3, false, false);
        }

        // ---- elementwise: h_new = 0.1*h + 0.9*tanh(pre); D layout: VGPR r -> M = r + 8*laneHi
        #pragma unroll
        for (int r = 0; r < 8; ++r) {
            int mm = mbase + r + laneHi * 8;
            float pre  = fast_tanh((acc0[r] + acc2[r]) + (acc1[r] + acc3[r]));
            float hnew = (1.0f - LEAK) * hstate[r] + LEAK * pre;
            hstate[r] = hnew;
            out[((size_t)mm * H_SZ + nglob) * T_SZ + t] = hnew;
            hnxt[(size_t)mm * H_SZ + nglob] = f2bf(hnew);
        }

        // ---- grid-wide barrier (monotonic counter, reset by init kernel each launch)
        __threadfence();
        __syncthreads();
        if (tid == 0) {
            __hip_atomic_fetch_add(barrier_cnt, 1u, __ATOMIC_RELEASE,
                                   __HIP_MEMORY_SCOPE_AGENT);
            unsigned target = (unsigned)(t + 1) * NWG;
            while (__hip_atomic_load(barrier_cnt, __ATOMIC_ACQUIRE,
                                     __HIP_MEMORY_SCOPE_AGENT) < target) {
                __builtin_amdgcn_s_sleep(1);
            }
        }
        __syncthreads();
        __threadfence();
    }
}

// ---------------- host launcher ----------------

extern "C" void kernel_launch(void* const* d_in, const int* in_sizes, int n_in,
                              void* d_out, int out_size, void* d_ws, size_t ws_size,
                              hipStream_t stream) {
    (void)in_sizes; (void)n_in; (void)out_size; (void)ws_size;

    const float* x     = (const float*)d_in[0];   // (B,C,T)
    const float* W_in  = (const float*)d_in[1];   // (H,C)
    const float* W_res = (const float*)d_in[2];   // (H,H)
    const float* b     = (const float*)d_in[3];   // (H)
    float* out = (float*)d_out;                   // (B,H,T)

    // Workspace layout (all 256B-aligned offsets), ~10.3 MB total:
    char* ws = (char*)d_ws;
    size_t o = 0;
    unsigned short* wres_bf = (unsigned short*)(ws + o); o += (size_t)H_SZ * H_SZ * 2;        // 2 MB
    unsigned short* win_bf  = (unsigned short*)(ws + o); o += (size_t)H_SZ * C_SZ * 2;        // 128 KB
    unsigned short* xbf     = (unsigned short*)(ws + o); o += (size_t)T_SZ * B_SZ * C_SZ * 2; // 8 MB
    unsigned short* hbf     = (unsigned short*)(ws + o); o += (size_t)2 * B_SZ * H_SZ * 2;    // 128 KB
    unsigned*       cnt     = (unsigned*)(ws + o);

    cvt_bf16_kernel<<<(H_SZ * H_SZ + 255) / 256, 256, 0, stream>>>(W_res, wres_bf, H_SZ * H_SZ);
    cvt_bf16_kernel<<<(H_SZ * C_SZ + 255) / 256, 256, 0, stream>>>(W_in, win_bf, H_SZ * C_SZ);
    xpose_x_kernel<<<(B_SZ * C_SZ * T_SZ + 255) / 256, 256, 0, stream>>>(x, xbf);
    init_state_kernel<<<(2 * B_SZ * H_SZ + 255) / 256, 256, 0, stream>>>(hbf, cnt);

    const size_t shmem = (size_t)(COLS_PER_WG * H_SZ + COLS_PER_WG * C_SZ) * sizeof(unsigned short); // 136 KB
    (void)hipFuncSetAttribute(reinterpret_cast<const void*>(esn_recurrence_kernel),
                              hipFuncAttributeMaxDynamicSharedMemorySize, (int)shmem);

    esn_recurrence_kernel<<<NWG, NTHREADS, shmem, stream>>>(
        xbf, wres_bf, win_bf, b, hbf, out, cnt);
}
